// RTNet_42133629174265
// MI455X (gfx1250) — matrix operand order repaired and checked by reading, not compiled, not run on hardware
//
#include <hip/hip_runtime.h>
#include <hip/hip_bf16.h>
#include <math.h>

typedef __attribute__((ext_vector_type(16))) _Float16 v16h;
typedef __attribute__((ext_vector_type(8)))  _Float16 v8h;
typedef __attribute__((ext_vector_type(2)))  _Float16 v2h;
typedef __attribute__((ext_vector_type(8)))  float    v8f;
typedef __attribute__((ext_vector_type(4)))  unsigned int u32x4;
typedef __attribute__((ext_vector_type(8)))  int      i32x8;
typedef __attribute__((ext_vector_type(4)))  int      i32x4;

#define B_   256
#define C_   64
#define L_   512
#define H_   128
#define G_   128
#define NC_  8
#define AT_  64
#define F1_  128
#define THRESH_ 0.02f

// ----------------------------------------------------------------------------
// WMMA GEMM: C[M,N] = act(A[M,K] @ B(K,N) + bias); f32 in memory, f16 fragments
// staged in double-buffered LDS, f32 accumulation via v_wmma_f32_16x16x32_f16.
// 128 threads = 4 waves; block tile 64x64; each wave computes a 16x64 strip.
// Pipeline: while WMMAs consume tile kt from LDS buffer p, the global loads for
// tile kt+1 are in flight into registers (8x b64 per thread, fully unrolled so
// they batch), stored to buffer 1-p after compute issue; one barrier per step.
// LDS layout: every fragment = two contiguous 16B chunks (ds_load_b128):
//   As row-major [m][k] (K pairs contiguous), Bt transposed [n][k].
// transB: B stored as (N,K) row-major -> B[k][n] = Bm[n*ldb + k]
// ----------------------------------------------------------------------------
__global__ __launch_bounds__(128)
void wmma_gemm_kernel(const float* __restrict__ A, int lda,
                      const float* __restrict__ Bm, int ldb, int transB,
                      float* __restrict__ Cout, int ldc,
                      int M, int N, int K,
                      const float* __restrict__ bias, int act)
{
    __shared__ _Float16 As[2][64][40];   // 32 halves used + 8 pad (80B rows)
    __shared__ _Float16 Bt[2][64][40];   // transposed tile: [n][k]
    const int tid  = threadIdx.x;
    const int wave = tid >> 5;
    const int lane = tid & 31;
    const int m0 = blockIdx.y * 64;
    const int n0 = blockIdx.x * 64;
    const int hi = lane >> 4;
    const int lo = lane & 15;

    // staging coordinates: i = tid + 128*u  ->  A/Bt(row-major): r=(tid>>4)+8u,
    // kp=tid&15 ; B(non-trans): c=tid&63, kp=(tid>>6)+2u
    const int akp = tid & 15, ar = tid >> 4;
    const int bc  = tid & 63, bk = tid >> 6;

    v8f zero = {0.f,0.f,0.f,0.f,0.f,0.f,0.f,0.f};
    v8f acc[4] = {zero, zero, zero, zero};

    float2 aR[8];
    float2 bR[8];
    float  bv0[8], bv1[8];

    // ---- prologue: load + store tile 0 ----
    #pragma unroll
    for (int u = 0; u < 8; ++u)
        aR[u] = *(const float2*)&A[(size_t)(m0 + ar + 8*u) * lda + 2*akp];
    if (transB) {
        #pragma unroll
        for (int u = 0; u < 8; ++u)
            bR[u] = *(const float2*)&Bm[(size_t)(n0 + ar + 8*u) * ldb + 2*akp];
    } else {
        #pragma unroll
        for (int u = 0; u < 8; ++u) {
            int kp = bk + 2*u;
            bv0[u] = Bm[(size_t)(2*kp)     * ldb + (n0 + bc)];
            bv1[u] = Bm[(size_t)(2*kp + 1) * ldb + (n0 + bc)];
        }
    }
    #pragma unroll
    for (int u = 0; u < 8; ++u) {
        v2h t; t[0] = (_Float16)aR[u].x; t[1] = (_Float16)aR[u].y;
        *(v2h*)&As[0][ar + 8*u][2*akp] = t;
    }
    if (transB) {
        #pragma unroll
        for (int u = 0; u < 8; ++u) {
            v2h t; t[0] = (_Float16)bR[u].x; t[1] = (_Float16)bR[u].y;
            *(v2h*)&Bt[0][ar + 8*u][2*akp] = t;
        }
    } else {
        #pragma unroll
        for (int u = 0; u < 8; ++u) {
            v2h t; t[0] = (_Float16)bv0[u]; t[1] = (_Float16)bv1[u];
            *(v2h*)&Bt[0][bc][2*(bk + 2*u)] = t;
        }
    }
    __syncthreads();

    const int NT = K >> 5;
    for (int kt = 0; kt < NT; ++kt) {
        const int p = kt & 1;
        const bool nxt = (kt + 1) < NT;
        const int k1 = (kt + 1) << 5;

        // issue global loads for next tile (latency hidden behind WMMAs)
        if (nxt) {
            #pragma unroll
            for (int u = 0; u < 8; ++u)
                aR[u] = *(const float2*)&A[(size_t)(m0 + ar + 8*u) * lda + (k1 + 2*akp)];
            if (transB) {
                #pragma unroll
                for (int u = 0; u < 8; ++u)
                    bR[u] = *(const float2*)&Bm[(size_t)(n0 + ar + 8*u) * ldb + (k1 + 2*akp)];
            } else {
                #pragma unroll
                for (int u = 0; u < 8; ++u) {
                    int kp = bk + 2*u;
                    bv0[u] = Bm[(size_t)(k1 + 2*kp)     * ldb + (n0 + bc)];
                    bv1[u] = Bm[(size_t)(k1 + 2*kp + 1) * ldb + (n0 + bc)];
                }
            }
        }

        // compute from buffer p
        // A fragment (ISA 7.12.2, 16-bit A 16x32):
        // lo lanes: K 0..7 | 16..23 ; hi lanes: K 8..15 | 24..31 -> 2x b128
        int mr = wave * 16 + lo;
        v8h a0 = *(const v8h*)&As[p][mr][hi ? 8 : 0];
        v8h a1 = *(const v8h*)&As[p][mr][hi ? 24 : 16];
        v16h a = __builtin_shufflevector(a0, a1, 0,1,2,3,4,5,6,7,8,9,10,11,12,13,14,15);
        #pragma unroll
        for (int nt = 0; nt < 4; ++nt) {
            // B fragment: lo lanes K 0..7 | 16..23 ; hi lanes K 8..15 | 24..31 -> 2x b128
            int nn = nt * 16 + lo;
            v8h b0 = *(const v8h*)&Bt[p][nn][hi ? 8 : 0];
            v8h b1 = *(const v8h*)&Bt[p][nn][hi ? 24 : 16];
            v16h b = __builtin_shufflevector(b0, b1, 0,1,2,3,4,5,6,7,8,9,10,11,12,13,14,15);
            acc[nt] = __builtin_amdgcn_wmma_f32_16x16x32_f16(
                false, a, false, b, (short)0, acc[nt], false, false);
        }

        // store next tile into the other buffer
        if (nxt) {
            #pragma unroll
            for (int u = 0; u < 8; ++u) {
                v2h t; t[0] = (_Float16)aR[u].x; t[1] = (_Float16)aR[u].y;
                *(v2h*)&As[1 - p][ar + 8*u][2*akp] = t;
            }
            if (transB) {
                #pragma unroll
                for (int u = 0; u < 8; ++u) {
                    v2h t; t[0] = (_Float16)bR[u].x; t[1] = (_Float16)bR[u].y;
                    *(v2h*)&Bt[1 - p][ar + 8*u][2*akp] = t;
                }
            } else {
                #pragma unroll
                for (int u = 0; u < 8; ++u) {
                    v2h t; t[0] = (_Float16)bv0[u]; t[1] = (_Float16)bv1[u];
                    *(v2h*)&Bt[1 - p][bc][2*(bk + 2*u)] = t;
                }
            }
        }
        __syncthreads();
    }

    #pragma unroll
    for (int nt = 0; nt < 4; ++nt) {
        #pragma unroll
        for (int r = 0; r < 8; ++r) {
            int row = m0 + wave * 16 + r + (hi ? 8 : 0);
            int col = n0 + nt * 16 + lo;
            float v = acc[nt][r];
            if (bias) v += bias[col];
            if (act == 1) v = fmaxf(v, 0.f);
            Cout[(size_t)row * ldc + col] = v;
        }
    }
}

// ---------------- GAT adjacency (algebraically fused) -----------------------
__global__ void gatw_vec_kernel(const float* __restrict__ W,
                                const float* __restrict__ a1,
                                const float* __restrict__ a2,
                                float* wa1, float* wa2)
{
    int j = blockIdx.x * blockDim.x + threadIdx.x;
    if (j >= L_) return;
    const float* wr = W + (size_t)j * L_;
    float s1 = 0.f, s2 = 0.f;
    for (int k = 0; k < L_; ++k) { s1 += wr[k] * a1[k]; s2 += wr[k] * a2[k]; }
    wa1[j] = s1; wa2[j] = s2;
}

__global__ void ha_kernel(const float* __restrict__ x,
                          const float* __restrict__ wa1,
                          const float* __restrict__ wa2,
                          float* ha1, float* ha2)
{
    int row = blockIdx.x * blockDim.x + threadIdx.x;   // b*C_ + i
    if (row >= B_ * C_) return;
    const float* xr = x + (size_t)row * L_;
    float s1 = 0.f, s2 = 0.f;
    for (int k = 0; k < L_; ++k) { float v = xr[k]; s1 += v * wa1[k]; s2 += v * wa2[k]; }
    ha1[row] = s1; ha2[row] = s2;
}

__global__ __launch_bounds__(64)
void adj_softmax_kernel(const float* __restrict__ ha1,
                        const float* __restrict__ ha2,
                        float* __restrict__ m)
{
    int row = blockIdx.x;                  // b*C_ + i
    int b = row >> 6;
    int j = threadIdx.x;
    __shared__ float red[64];
    float e = ha1[row] + ha2[b * C_ + j];
    e = (e > 0.f) ? e : 0.2f * e;          // leaky 0.2
    red[j] = e; __syncthreads();
    for (int s = 32; s > 0; s >>= 1) { if (j < s) red[j] = fmaxf(red[j], red[j + s]); __syncthreads(); }
    float mx = red[0]; __syncthreads();
    float ex = expf(e - mx);
    red[j] = ex; __syncthreads();
    for (int s = 32; s > 0; s >>= 1) { if (j < s) red[j] += red[j + s]; __syncthreads(); }
    float p = ex / red[0];
    m[(size_t)row * C_ + j] = (p > THRESH_) ? 1.f : 0.f;
}

// ---------------- GCN ----------------
__global__ void dinv_kernel(const float* __restrict__ m, float* dinv, int n)
{
    int b = blockIdx.x; int i = threadIdx.x;
    if (i >= n) return;
    const float* mr = m + ((size_t)b * n + i) * n;
    float s = 1.f;                          // +I on diagonal
    for (int j = 0; j < n; ++j) s += mr[j];
    dinv[b * n + i] = rsqrtf(s);
}

// One block per batch sample; xw tile staged into LDS via the Tensor Data
// Mover (TDM) when available (tensor_load_to_lds + s_wait_tensorcnt), with a
// plain cooperative-copy fallback.
__global__ __launch_bounds__(256)
void gcn_combine_kernel(const float* __restrict__ xw, const float* __restrict__ m,
                        const float* __restrict__ dinv, const float* __restrict__ bias,
                        float* __restrict__ outp, int n)
{
    __shared__ float tile[64 * H_];        // up to 32KB
    int b = blockIdx.x;
    const float* src = xw + (size_t)b * n * H_;
#if __has_builtin(__builtin_amdgcn_tensor_load_to_lds)
    if (threadIdx.x == 0) {
        unsigned long long ga = (unsigned long long)(uintptr_t)src;
        unsigned lds_addr = (unsigned)(uintptr_t)&tile[0];
        u32x4 g0;
        g0[0] = 1u;                                        // count=1, user mode
        g0[1] = lds_addr;                                  // lds_addr
        g0[2] = (unsigned)(ga & 0xffffffffu);              // global addr lo
        g0[3] = (unsigned)((ga >> 32) & 0x01ffffffu) | (2u << 30);  // addr hi | type=2
        i32x8 g1;
        g1[0] = (int)(2u << 16);                           // data_size=4B, mask=0
        g1[1] = (int)(((unsigned)H_) << 16);               // tensor_dim0=128 (lo16)
        g1[2] = (int)(((unsigned)n & 0xffffu) << 16);      // dim0 hi=0 | tensor_dim1=n lo16
        g1[3] = (int)(((unsigned)H_) << 16);               // dim1 hi=0 | tile_dim0=128
        g1[4] = (int)(unsigned)n;                          // tile_dim1=n, tile_dim2=0
        g1[5] = H_;                                        // tensor_dim0_stride lo32 = 128
        g1[6] = 0;
        g1[7] = 0;
        i32x4 gz; gz[0] = 0; gz[1] = 0; gz[2] = 0; gz[3] = 0;
#if __clang_major__ >= 23
        i32x8 gz8; for (int q = 0; q < 8; ++q) gz8[q] = 0;
        __builtin_amdgcn_tensor_load_to_lds(g0, g1, gz, gz, gz8, 0);
#else
        __builtin_amdgcn_tensor_load_to_lds(g0, g1, gz, gz, 0);
#endif
        __builtin_amdgcn_s_wait_tensorcnt(0);
    }
    __syncthreads();
#else
    for (int i = threadIdx.x; i < n * H_; i += 256) tile[i] = src[i];
    __syncthreads();
#endif
    for (int idx = threadIdx.x; idx < n * H_; idx += 256) {
        int i = idx >> 7, f = idx & 127;
        const float* mr = m + ((size_t)b * n + i) * n;
        float di = dinv[b * n + i];
        float acc = 0.f;
        for (int j = 0; j < n; ++j) {
            float a = mr[j] + ((i == j) ? 1.f : 0.f);
            acc += di * a * dinv[b * n + j] * tile[j * H_ + f];
        }
        outp[((size_t)b * n + i) * H_ + f] = fmaxf(acc + bias[f], 0.f);
    }
}

// ---------------- SAGPool ----------------
__global__ void sag_dots_kernel(const float* __restrict__ x, const float* __restrict__ w1,
                                const float* __restrict__ w2, float* d1, float* d2, int n)
{
    int row = blockIdx.x * blockDim.x + threadIdx.x;
    if (row >= B_ * n) return;
    const float* xr = x + (size_t)row * H_;
    float s1 = 0.f, s2 = 0.f;
    for (int f = 0; f < H_; ++f) { float v = xr[f]; s1 += v * w1[f]; s2 += v * w2[f]; }
    d1[row] = s1; d2[row] = s2;
}

__global__ void sag_score_kernel(const float* __restrict__ d1, const float* __restrict__ d2,
                                 const float* __restrict__ m, const float* __restrict__ bscal,
                                 float* s, int n)
{
    int row = blockIdx.x * blockDim.x + threadIdx.x;
    if (row >= B_ * n) return;
    int b = row / n; int i = row % n;
    const float* mr = m + ((size_t)b * n + i) * n;
    float acc = d1[row] + bscal[0];
    for (int j = 0; j < n; ++j) acc += mr[j] * d2[b * n + j];
    s[row] = acc;
}

__global__ void topk_kernel(const float* __restrict__ s, int n, int k,
                            int* idx, float* gate)
{
    int b = blockIdx.x;
    if (threadIdx.x != 0) return;
    float sc[64];
    for (int i = 0; i < n; ++i) sc[i] = s[b * n + i];
    for (int kk = 0; kk < k; ++kk) {
        int best = 0; float bv = sc[0];
        for (int i = 1; i < n; ++i) if (sc[i] > bv) { bv = sc[i]; best = i; }
        idx[b * k + kk] = best;
        gate[b * k + kk] = tanhf(bv);
        sc[best] = -3.4e38f;
    }
}

__global__ __launch_bounds__(128)
void sag_gather_kernel(const float* __restrict__ x, const float* __restrict__ m,
                       const int* __restrict__ idx, const float* __restrict__ gate,
                       float* xg, float* mg, int n, int k)
{
    int b = blockIdx.x;
    for (int i = threadIdx.x; i < k * H_; i += blockDim.x) {
        int kk = i / H_, f = i % H_;
        xg[((size_t)b * k + kk) * H_ + f] =
            x[((size_t)b * n + idx[b * k + kk]) * H_ + f] * gate[b * k + kk];
    }
    for (int i = threadIdx.x; i < k * k; i += blockDim.x) {
        int a = i / k, c2 = i % k;
        mg[((size_t)b * k + a) * k + c2] =
            m[((size_t)b * n + idx[b * k + a]) * n + idx[b * k + c2]];
    }
}

__global__ __launch_bounds__(128)
void readout_kernel(const float* __restrict__ g, int n, float* outp, int first)
{
    int b = blockIdx.x; int f = threadIdx.x;   // 128
    float mx = -3.4e38f, sm = 0.f;
    for (int i = 0; i < n; ++i) {
        float v = g[((size_t)b * n + i) * H_ + f];
        mx = fmaxf(mx, v); sm += v;
    }
    float mean = sm / n;
    if (first) { outp[(size_t)b * 2 * H_ + f] = mx; outp[(size_t)b * 2 * H_ + H_ + f] = mean; }
    else       { outp[(size_t)b * 2 * H_ + f] += mx; outp[(size_t)b * 2 * H_ + H_ + f] += mean; }
}

// ---------------- GRU recurrence (input gates precomputed by WMMA GEMM) -----
__global__ __launch_bounds__(256)
void gru_kernel(const float* __restrict__ gi,    // [B_][C_][3G_]
                const float* __restrict__ Whh,   // [3G_][G_]
                const float* __restrict__ bhh,   // [3G_]
                float* __restrict__ xgru)        // [B_][G_]
{
    __shared__ float h[C_ * G_];
    for (int i = threadIdx.x; i < C_ * G_; i += 256) h[i] = 0.f;
    __syncthreads();
    float hn_loc[32];
    for (int t = 0; t < B_; ++t) {
        const float* git = gi + (size_t)t * C_ * 3 * G_;
        int slot = 0;
        for (int p = threadIdx.x; p < C_ * G_; p += 256, ++slot) {
            int c = p >> 7, g = p & 127;
            const float* hc = h + c * G_;
            float hr = bhh[g], hz = bhh[G_ + g], hnv = bhh[2 * G_ + g];
            const float* wr = Whh + (size_t)g * G_;
            const float* wz = Whh + (size_t)(G_ + g) * G_;
            const float* wn = Whh + (size_t)(2 * G_ + g) * G_;
            for (int j = 0; j < G_; ++j) {
                float hv = hc[j];
                hr += hv * wr[j]; hz += hv * wz[j]; hnv += hv * wn[j];
            }
            float ir  = git[c * 3 * G_ + g];
            float iz  = git[c * 3 * G_ + G_ + g];
            float inn = git[c * 3 * G_ + 2 * G_ + g];
            float rg = 1.f / (1.f + expf(-(ir + hr)));
            float zg = 1.f / (1.f + expf(-(iz + hz)));
            float ng = tanhf(inn + rg * hnv);
            hn_loc[slot] = (1.f - zg) * ng + zg * hc[g];
        }
        __syncthreads();
        slot = 0;
        for (int p = threadIdx.x; p < C_ * G_; p += 256, ++slot) h[p] = hn_loc[slot];
        __syncthreads();
        if (threadIdx.x < G_) {
            float s = 0.f;
            for (int c = 0; c < C_; ++c) s += h[c * G_ + threadIdx.x];
            xgru[(size_t)t * G_ + threadIdx.x] = s * (1.f / C_);
        }
        __syncthreads();
    }
}

// ---------------- dilated conv branch ----------------
__global__ __launch_bounds__(128)
void conv1d_relu_kernel(const float* __restrict__ in, int Cin, int Lin,
                        const float* __restrict__ W, const float* __restrict__ bias,
                        float* __restrict__ outp, int Lout, int Kk, int dil, int pad)
{
    int b = blockIdx.x, f = blockIdx.y;
    const float* xb = in + (size_t)b * Cin * Lin;
    const float* wf = W + (size_t)f * Cin * Kk;
    for (int o = threadIdx.x; o < Lout; o += blockDim.x) {
        float acc = bias[f];
        for (int c = 0; c < Cin; ++c) {
            const float* xc = xb + (size_t)c * Lin;
            const float* wc = wf + c * Kk;
            for (int k = 0; k < Kk; ++k) {
                int ix = o + dil * k - pad;
                if (ix >= 0 && ix < Lin) acc += xc[ix] * wc[k];
            }
        }
        outp[((size_t)b * gridDim.y + f) * Lout + o] = fmaxf(acc, 0.f);
    }
}

__global__ __launch_bounds__(128)
void conv1d_relu_mean_kernel(const float* __restrict__ in, int Cin, int Lin,
                             const float* __restrict__ W, const float* __restrict__ bias,
                             float* __restrict__ outp, int Lout, int Kk, int dil, int pad)
{
    int b = blockIdx.x, f = blockIdx.y;
    __shared__ float red[128];
    const float* xb = in + (size_t)b * Cin * Lin;
    const float* wf = W + (size_t)f * Cin * Kk;
    float total = 0.f;
    for (int o = threadIdx.x; o < Lout; o += blockDim.x) {
        float acc = bias[f];
        for (int c = 0; c < Cin; ++c) {
            const float* xc = xb + (size_t)c * Lin;
            const float* wc = wf + c * Kk;
            for (int k = 0; k < Kk; ++k) {
                int ix = o + dil * k - pad;
                if (ix >= 0 && ix < Lin) acc += xc[ix] * wc[k];
            }
        }
        total += fmaxf(acc, 0.f);
    }
    red[threadIdx.x] = total; __syncthreads();
    for (int s = 64; s > 0; s >>= 1) { if (threadIdx.x < s) red[threadIdx.x] += red[threadIdx.x + s]; __syncthreads(); }
    if (threadIdx.x == 0) outp[(size_t)b * gridDim.y + f] = red[0] / Lout;
}

// ---------------- head ----------------
__global__ void feat_kernel(const float* xconv, const float* xgru, const float* xgnn,
                            float* feat)
{
    int b = blockIdx.x; int j = threadIdx.x;   // 512
    float v;
    if (j < 128)      v = xconv[b * 128 + j];
    else if (j < 256) v = xgru[b * 128 + (j - 128)];
    else              v = xgnn[b * 256 + (j - 256)];
    feat[(size_t)b * 512 + j] = v;
}

__global__ __launch_bounds__(256)
void bn_leaky_kernel(float* hm, const float* g, const float* b)
{
    int f = blockIdx.x;      // 256 features
    int r = threadIdx.x;     // 256 rows
    __shared__ float red[256];
    float v = hm[(size_t)r * 256 + f];
    red[r] = v; __syncthreads();
    for (int s = 128; s > 0; s >>= 1) { if (r < s) red[r] += red[r + s]; __syncthreads(); }
    float mean = red[0] / 256.f; __syncthreads();
    float dv = v - mean;
    red[r] = dv * dv; __syncthreads();
    for (int s = 128; s > 0; s >>= 1) { if (r < s) red[r] += red[r + s]; __syncthreads(); }
    float var = red[0] / 256.f;
    float y = g[f] * dv * rsqrtf(var + 1e-5f) + b[f];
    hm[(size_t)r * 256 + f] = (y > 0.f) ? y : 0.01f * y;
}

__global__ void trainmask_kernel(const int* idx_train, int ntrain, float* tm)
{
    int t = threadIdx.x;
    if (t < B_) tm[t] = 0.f;
    __syncthreads();
    if (t < ntrain) tm[idx_train[t]] = 1.f;
}

__global__ __launch_bounds__(256)
void att_proto_kernel(const float* __restrict__ z, const int* __restrict__ labels,
                      const float* __restrict__ tm,
                      const float* __restrict__ W1, const float* __restrict__ b1,
                      const float* __restrict__ W2, const float* __restrict__ b2,
                      float* __restrict__ proto)
{
    int c = blockIdx.x;
    int n = threadIdx.x;    // 256 = B_
    __shared__ float red[256];
    __shared__ float wbuf[256];
    float logit = b2[c];
    const float* W1c = W1 + (size_t)c * 128 * AT_;
    for (int a = 0; a < AT_; ++a) {
        float s = b1[c * AT_ + a];
        for (int d = 0; d < 128; ++d)
            s += z[(size_t)n * 128 + d] * W1c[d * AT_ + a];
        logit += tanhf(s) * W2[c * AT_ + a];
    }
    bool ok = (tm[n] > 0.5f) && (labels[n] == c);
    logit = ok ? logit : -1e30f;
    red[n] = logit; __syncthreads();
    for (int s = 128; s > 0; s >>= 1) { if (n < s) red[n] = fmaxf(red[n], red[n + s]); __syncthreads(); }
    float mx = red[0]; __syncthreads();
    float ex = expf(logit - mx);
    red[n] = ex; __syncthreads();
    for (int s = 128; s > 0; s >>= 1) { if (n < s) red[n] += red[n + s]; __syncthreads(); }
    float sum = red[0];
    wbuf[n] = ex / sum;
    __syncthreads();
    if (n < 128) {
        float p = 0.f;
        for (int q = 0; q < 256; ++q) p += wbuf[q] * z[(size_t)q * 128 + n];
        proto[c * 128 + n] = p;
    }
}

__global__ void dists_kernel(const float* __restrict__ z, const float* __restrict__ proto,
                             float* __restrict__ outp)
{
    int n = blockIdx.x;
    int c = threadIdx.x;
    if (c < NC_) {
        float d2 = 0.f;
        for (int d = 0; d < 128; ++d) {
            float df = z[(size_t)n * 128 + d] - proto[c * 128 + d];
            d2 += df * df;
        }
        d2 = fmaxf(d2, 1e-12f);
        outp[n * NC_ + c] = expf(-0.5f * sqrtf(d2));
    }
}

__global__ void protodist_kernel(const float* __restrict__ proto, float* __restrict__ outp)
{
    __shared__ float red[64];
    int p = threadIdx.x;
    int i = p >> 3, j = p & 7;
    float d2 = 0.f;
    for (int d = 0; d < 128; ++d) {
        float df = proto[i * 128 + d] - proto[j * 128 + d];
        d2 += df * df;
    }
    d2 = fmaxf(d2, 1e-12f);
    red[p] = expf(-0.5f * sqrtf(d2));
    __syncthreads();
    for (int s = 32; s > 0; s >>= 1) { if (p < s) red[p] += red[p + s]; __syncthreads(); }
    if (p == 0) outp[0] = red[0] / (NC_ * (NC_ - 1) / 2.0f);
}

// ----------------------------------------------------------------------------
extern "C" void kernel_launch(void* const* d_in, const int* in_sizes, int n_in,
                              void* d_out, int out_size, void* d_ws, size_t ws_size,
                              hipStream_t stream)
{
    (void)n_in; (void)out_size; (void)ws_size;
    const float* x         = (const float*)d_in[0];
    const int*   labels    = (const int*)d_in[1];
    const int*   idx_train = (const int*)d_in[2];
    const float* gat_W     = (const float*)d_in[3];
    const float* gat_a1    = (const float*)d_in[4];
    const float* gat_a2    = (const float*)d_in[5];
    const float* gcn1_W    = (const float*)d_in[6];
    const float* gcn1_b    = (const float*)d_in[7];
    const float* gcn2_W    = (const float*)d_in[8];
    const float* gcn2_b    = (const float*)d_in[9];
    const float* gcn3_W    = (const float*)d_in[10];
    const float* gcn3_b    = (const float*)d_in[11];
    const float* pool1_w1  = (const float*)d_in[12];
    const float* pool1_w2  = (const float*)d_in[13];
    const float* pool1_b   = (const float*)d_in[14];
    const float* pool2_w1  = (const float*)d_in[15];
    const float* pool2_w2  = (const float*)d_in[16];
    const float* pool2_b   = (const float*)d_in[17];
    const float* pool3_w1  = (const float*)d_in[18];
    const float* pool3_w2  = (const float*)d_in[19];
    const float* pool3_b   = (const float*)d_in[20];
    const float* gru_Wih   = (const float*)d_in[21];
    const float* gru_Whh   = (const float*)d_in[22];
    const float* gru_bih   = (const float*)d_in[23];
    const float* gru_bhh   = (const float*)d_in[24];
    const float* conv1_W   = (const float*)d_in[25];
    const float* conv1_b   = (const float*)d_in[26];
    const float* conv2_W   = (const float*)d_in[27];
    const float* conv2_b   = (const float*)d_in[28];
    const float* conv3_W   = (const float*)d_in[29];
    const float* conv3_b   = (const float*)d_in[30];
    const float* fc0_W     = (const float*)d_in[31];
    const float* fc0_b     = (const float*)d_in[32];
    const float* bn0_g     = (const float*)d_in[33];
    const float* bn0_b     = (const float*)d_in[34];
    const float* fc1_W     = (const float*)d_in[35];
    const float* fc1_b     = (const float*)d_in[36];
    const float* att_W1    = (const float*)d_in[37];
    const float* att_b1    = (const float*)d_in[38];
    const float* att_W2    = (const float*)d_in[39];
    const float* att_b2    = (const float*)d_in[40];

    float* out = (float*)d_out;

    float* base = (float*)d_ws;
    size_t off = 0;
    auto alloc = [&](size_t nf) -> float* { float* p = base + off; off += nf; return p; };

    float* wa1   = alloc(L_);
    float* wa2   = alloc(L_);
    float* ha1   = alloc(B_ * C_);
    float* ha2   = alloc(B_ * C_);
    float* m0    = alloc((size_t)B_ * 64 * 64);
    float* dinv  = alloc(B_ * 64);
    float* xw    = alloc((size_t)B_ * 64 * H_);
    float* g1    = alloc((size_t)B_ * 64 * H_);
    float* gp1   = alloc((size_t)B_ * 32 * H_);
    float* m1    = alloc((size_t)B_ * 32 * 32);
    float* g2    = alloc((size_t)B_ * 32 * H_);
    float* gp2   = alloc((size_t)B_ * 16 * H_);
    float* m2    = alloc((size_t)B_ * 16 * 16);
    float* g3    = alloc((size_t)B_ * 16 * H_);
    float* gp3   = alloc((size_t)B_ * 8 * H_);
    float* sbuf  = alloc(B_ * 64);
    float* dd1   = alloc(B_ * 64);
    float* dd2   = alloc(B_ * 64);
    float* gate  = alloc(B_ * 32);
    int*   idxb  = (int*)alloc(B_ * 32);
    float* xgnn  = alloc(B_ * 256);
    float* gi    = alloc((size_t)B_ * C_ * 3 * G_);
    float* xgru  = alloc((size_t)B_ * G_);
    float* y1    = alloc((size_t)B_ * F1_ * 519);
    float* y2    = alloc((size_t)B_ * F1_ * 527);
    float* xconv = alloc(B_ * F1_);
    float* feat  = alloc((size_t)B_ * 512);
    float* hm    = alloc((size_t)B_ * 256);
    float* zbuf  = alloc((size_t)B_ * 128);
    float* tm    = alloc(B_);
    float* proto = alloc(NC_ * 128);

    // --- learned adjacency (fused: ha = x @ (gat_W @ a)) ---
    gatw_vec_kernel<<<4, 128, 0, stream>>>(gat_W, gat_a1, gat_a2, wa1, wa2);
    ha_kernel<<<128, 128, 0, stream>>>(x, wa1, wa2, ha1, ha2);
    adj_softmax_kernel<<<B_ * C_, 64, 0, stream>>>(ha1, ha2, m0);

    // --- GCN1 + pool1 ---
    wmma_gemm_kernel<<<dim3(H_/64, (B_*C_)/64), 128, 0, stream>>>(
        x, L_, gcn1_W, H_, 0, xw, H_, B_*C_, H_, L_, nullptr, 0);
    dinv_kernel<<<B_, 64, 0, stream>>>(m0, dinv, 64);
    gcn_combine_kernel<<<B_, 256, 0, stream>>>(xw, m0, dinv, gcn1_b, g1, 64);
    sag_dots_kernel<<<(B_*64+127)/128, 128, 0, stream>>>(g1, pool1_w1, pool1_w2, dd1, dd2, 64);
    sag_score_kernel<<<(B_*64+127)/128, 128, 0, stream>>>(dd1, dd2, m0, pool1_b, sbuf, 64);
    topk_kernel<<<B_, 32, 0, stream>>>(sbuf, 64, 32, idxb, gate);
    sag_gather_kernel<<<B_, 128, 0, stream>>>(g1, m0, idxb, gate, gp1, m1, 64, 32);
    readout_kernel<<<B_, 128, 0, stream>>>(gp1, 32, xgnn, 1);

    // --- GCN2 + pool2 ---
    wmma_gemm_kernel<<<dim3(H_/64, (B_*32)/64), 128, 0, stream>>>(
        gp1, H_, gcn2_W, H_, 0, xw, H_, B_*32, H_, H_, nullptr, 0);
    dinv_kernel<<<B_, 32, 0, stream>>>(m1, dinv, 32);
    gcn_combine_kernel<<<B_, 256, 0, stream>>>(xw, m1, dinv, gcn2_b, g2, 32);
    sag_dots_kernel<<<(B_*32+127)/128, 128, 0, stream>>>(g2, pool2_w1, pool2_w2, dd1, dd2, 32);
    sag_score_kernel<<<(B_*32+127)/128, 128, 0, stream>>>(dd1, dd2, m1, pool2_b, sbuf, 32);
    topk_kernel<<<B_, 32, 0, stream>>>(sbuf, 32, 16, idxb, gate);
    sag_gather_kernel<<<B_, 128, 0, stream>>>(g2, m1, idxb, gate, gp2, m2, 32, 16);
    readout_kernel<<<B_, 128, 0, stream>>>(gp2, 16, xgnn, 0);

    // --- GCN3 + pool3 ---
    wmma_gemm_kernel<<<dim3(H_/64, (B_*16)/64), 128, 0, stream>>>(
        gp2, H_, gcn3_W, H_, 0, xw, H_, B_*16, H_, H_, nullptr, 0);
    dinv_kernel<<<B_, 16, 0, stream>>>(m2, dinv, 16);
    gcn_combine_kernel<<<B_, 256, 0, stream>>>(xw, m2, dinv, gcn3_b, g3, 16);
    sag_dots_kernel<<<(B_*16+127)/128, 128, 0, stream>>>(g3, pool3_w1, pool3_w2, dd1, dd2, 16);
    sag_score_kernel<<<(B_*16+127)/128, 128, 0, stream>>>(dd1, dd2, m2, pool3_b, sbuf, 16);
    topk_kernel<<<B_, 32, 0, stream>>>(sbuf, 16, 8, idxb, gate);
    sag_gather_kernel<<<B_, 128, 0, stream>>>(g3, m2, idxb, gate, gp3, m1 /*scratch*/, 16, 8);
    readout_kernel<<<B_, 128, 0, stream>>>(gp3, 8, xgnn, 0);

    // --- GRU: input gates as one big WMMA GEMM, then tiny LDS-resident scan ---
    wmma_gemm_kernel<<<dim3((3*G_)/64, (B_*C_)/64), 128, 0, stream>>>(
        x, L_, gru_Wih, L_, 1, gi, 3*G_, B_*C_, 3*G_, L_, gru_bih, 0);
    gru_kernel<<<1, 256, 0, stream>>>(gi, gru_Whh, gru_bhh, xgru);

    // --- dilated conv branch ---
    conv1d_relu_kernel<<<dim3(B_, F1_), 128, 0, stream>>>(x,  C_,  L_,  conv1_W, conv1_b, y1, 519, 8, 1, 7);
    conv1d_relu_kernel<<<dim3(B_, F1_), 128, 0, stream>>>(y1, F1_, 519, conv2_W, conv2_b, y2, 527, 5, 2, 8);
    conv1d_relu_mean_kernel<<<dim3(B_, F1_), 128, 0, stream>>>(y2, F1_, 527, conv3_W, conv3_b, xconv, 535, 3, 4, 8);

    // --- head ---
    feat_kernel<<<B_, 512, 0, stream>>>(xconv, xgru, xgnn, feat);
    wmma_gemm_kernel<<<dim3(256/64, B_/64), 128, 0, stream>>>(
        feat, 512, fc0_W, 256, 0, hm, 256, B_, 256, 512, fc0_b, 0);
    bn_leaky_kernel<<<256, 256, 0, stream>>>(hm, bn0_g, bn0_b);
    wmma_gemm_kernel<<<dim3(128/64, B_/64), 128, 0, stream>>>(
        hm, 256, fc1_W, 128, 0, zbuf, 128, B_, 128, 256, fc1_b, 0);

    trainmask_kernel<<<1, 256, 0, stream>>>(idx_train, in_sizes[2], tm);
    att_proto_kernel<<<NC_, 256, 0, stream>>>(zbuf, labels, tm, att_W1, att_b1, att_W2, att_b2, proto);
    dists_kernel<<<B_, 32, 0, stream>>>(zbuf, proto, out);
    protodist_kernel<<<1, 64, 0, stream>>>(proto, out + B_ * NC_);
}
